// DCUnetBlock_31447750541758
// MI455X (gfx1250) — matrix operand order repaired and checked by reading, not compiled, not run on hardware
//
#include <hip/hip_runtime.h>
#include <hip/hip_bf16.h>

typedef __attribute__((ext_vector_type(16))) __bf16 v16bf;
typedef __attribute__((ext_vector_type(8)))  __bf16 v8bf;
typedef __attribute__((ext_vector_type(8)))  float  v8f;
typedef __attribute__((ext_vector_type(4))) unsigned int u32x4;
typedef __attribute__((ext_vector_type(8))) int i32x8;
typedef __attribute__((ext_vector_type(4))) int i32x4;
typedef __attribute__((address_space(1))) i32x4 g_i32x4;   // global-space vec4
typedef __attribute__((address_space(3))) i32x4 l_i32x4;   // LDS-space vec4

#if __has_builtin(__builtin_amdgcn_tensor_load_to_lds) && \
    __has_builtin(__builtin_amdgcn_s_wait_tensorcnt)
#define HAVE_TDM 1
#endif
#if __has_builtin(__builtin_amdgcn_global_load_async_to_lds_b128) && \
    __has_builtin(__builtin_amdgcn_s_wait_asynccnt)
#define HAVE_ASYNC 1
#endif

// ---------------------------------------------------------------------------
// Weight prep: convert to bf16 in WMMA-B-matrix friendly [N][K] row-major.
// ---------------------------------------------------------------------------
__global__ void k_prep_pw(const float* __restrict__ w, __bf16* __restrict__ Bm) {
  int i = blockIdx.x * blockDim.x + threadIdx.x;
  if (i < 64 * 64) Bm[i] = (__bf16)w[i];           // pw_w is already [o][c]
}

__global__ void k_prep_dcn(const float* __restrict__ w, __bf16* __restrict__ Bm) {
  // dcn_w: [o=64][c=64][k=9]  ->  Bm: [n=o][kk = k*64 + c]
  int i = blockIdx.x * blockDim.x + threadIdx.x;
  if (i < 64 * 576) {
    int n = i / 576, kk = i % 576, k = kk / 64, c = kk % 64;
    Bm[i] = (__bf16)w[(n * 64 + c) * 9 + k];
  }
}

// ---------------------------------------------------------------------------
// Fused depthwise 3x3 (+bias) -> bf16 LDS tile -> WMMA pointwise 1x1 (+bias)
// One block = 16x16 pixel tile, all 64 channels.  GEMM: M=256, N=64, K=64.
// The 8KB pointwise B-matrix is DMA'd to LDS by the Tensor Data Mover while
// phase 1 (depthwise) computes; s_wait_tensorcnt gates phase 2.
// ---------------------------------------------------------------------------
__global__ __launch_bounds__(256) void k_dw_pw(
    const float* __restrict__ x, const float* __restrict__ dw_w,
    const float* __restrict__ dw_b, const __bf16* __restrict__ Bpw,
    const float* __restrict__ pw_b, float* __restrict__ out) {
  constexpr int C = 64, H = 256, W = 256;
  __shared__ alignas(32) __bf16 As[256 * C];        // 32 KB
  __shared__ alignas(32) __bf16 Bs[64 * C];         // 8 KB
  const int x0 = blockIdx.x * 16, y0 = blockIdx.y * 16, b = blockIdx.z;
  const int t = threadIdx.x;
  const int gy = y0 + (t >> 4), gx = x0 + (t & 15);

#ifdef HAVE_TDM
  // Tensor Data Mover: 1-row tile, 4096 elements x 2 bytes, global -> LDS.
  if (t == 0) {
    const unsigned lds_off =
        (unsigned)(size_t)(__attribute__((address_space(3))) const __bf16*)&Bs[0];
    const unsigned long long ga = (unsigned long long)(size_t)Bpw;
    u32x4 g0 = {0u, 0u, 0u, 0u};
    g0[0] = 1u;                                     // count=1 (valid descriptor)
    g0[1] = lds_off;                                // lds_addr
    g0[2] = (unsigned)ga;                           // global_addr[31:0]
    g0[3] = (unsigned)(ga >> 32) | (2u << 30);      // global_addr[56:32] | type=2
    i32x8 g1 = {0, 0, 0, 0, 0, 0, 0, 0};
    g1[0] = 0x10000;                                // data_size=1 (2 bytes)
    g1[1] = (int)(0x1000u << 16);                   // tensor_dim0=4096 (bits 63:48)
    g1[2] = (int)(1u << 16);                        // tensor_dim1=1 (bits 95:80)
    g1[3] = (int)(0x1000u << 16);                   // tile_dim0=4096 (bits 127:112)
    g1[4] = 1;                                      // tile_dim1=1
    g1[5] = 4096;                                   // tensor_dim0_stride
    i32x4 g2 = {0, 0, 0, 0};
    i32x4 g3 = {0, 0, 0, 0};
    i32x8 g4 = {0, 0, 0, 0, 0, 0, 0, 0};
    __builtin_amdgcn_tensor_load_to_lds(g0, g1, g2, g3, g4, 0);
  }
#else
  for (int i = t; i < 64 * C; i += 256) Bs[i] = Bpw[i];
#endif

  // phase 1: depthwise conv, this thread's pixel, all channels
  for (int c = 0; c < C; ++c) {
    const float* wc = dw_w + c * 9;
    const float* xc = x + (size_t)(b * C + c) * H * W;
    float acc = dw_b[c];
#pragma unroll
    for (int dy = -1; dy <= 1; ++dy) {
      const int yy = gy + dy;
      const bool vy = (yy >= 0) & (yy < H);
#pragma unroll
      for (int dx = -1; dx <= 1; ++dx) {
        const int xx = gx + dx;
        const bool v = vy & (xx >= 0) & (xx < W);
        const float xv = v ? xc[yy * W + xx] : 0.0f;
        acc += xv * wc[(dy + 1) * 3 + (dx + 1)];
      }
    }
    As[t * C + c] = (__bf16)acc;
  }
#ifdef HAVE_TDM
  __builtin_amdgcn_s_wait_tensorcnt(0);             // B-matrix DMA complete
#endif
  __syncthreads();

  // phase 2: WMMA pointwise GEMM (uniform control flow; EXEC all ones)
  const int wave = t >> 5, lane = t & 31;
  const int lrow = lane & 15;
  const int khalf = (lane >> 4) << 3;               // unused: 0 or 8
  const int kh16  = (lane >> 4) << 4;               // A and B: 0 or 16
  for (int tile = wave; tile < 64; tile += 8) {
    const int mt = tile >> 2, nt = tile & 3;
    const int arow = mt * 16 + lrow;
    const int bcol = nt * 16 + lrow;
    v8f acc = {};
#pragma unroll
    for (int k0 = 0; k0 < C; k0 += 32) {
      v16bf a = *(const v16bf*)(&As[arow * C + k0 + kh16]);
      v16bf bm = *(const v16bf*)(&Bs[bcol * C + k0 + kh16]);
      acc = __builtin_amdgcn_wmma_f32_16x16x32_bf16(false, a, false, bm,
                                                    (short)0, acc, false, false);
    }
    const int n = nt * 16 + lrow;
    const int mbase = mt * 16 + ((lane >> 4) << 3);
    const float bias = pw_b[n];
#pragma unroll
    for (int r = 0; r < 8; ++r) {
      const int m = mbase + r;
      out[((size_t)(b * C + n) * H + (y0 + (m >> 4))) * W + (x0 + (m & 15))] =
          acc[r] + bias;
    }
  }
}

// ---------------------------------------------------------------------------
// Offset conv: 3x3, Cin=64, Cout=18 (direct; uniform weight addresses -> SMEM)
// One thread = one (b,y,x), all 18 outputs.
// ---------------------------------------------------------------------------
__global__ __launch_bounds__(256) void k_off(
    const float* __restrict__ h, const float* __restrict__ ow,
    const float* __restrict__ ob, float* __restrict__ off) {
  constexpr int C = 64, H = 256, W = 256, CO = 18;
  const int p = blockIdx.x * blockDim.x + threadIdx.x;  // [0, B*H*W)
  const int b = p >> 16, rem = p & 65535;
  const int y = rem >> 8, xcol = rem & 255;
  float acc[CO];
#pragma unroll
  for (int o = 0; o < CO; ++o) acc[o] = ob[o];
  for (int c = 0; c < C; ++c) {
    const float* hc = h + (size_t)(b * C + c) * H * W;
    if (c + 1 < C) __builtin_prefetch(hc + H * W + y * W + xcol, 0, 1);
#pragma unroll
    for (int dy = -1; dy <= 1; ++dy) {
      const int yy = y + dy;
      const bool vy = (yy >= 0) & (yy < H);
#pragma unroll
      for (int dx = -1; dx <= 1; ++dx) {
        const int xx = xcol + dx;
        const bool v = vy & (xx >= 0) & (xx < W);
        const float hv = v ? hc[yy * W + xx] : 0.0f;
        const int tap = (dy + 1) * 3 + (dx + 1);
#pragma unroll
        for (int o = 0; o < CO; ++o) acc[o] += hv * ow[(o * C + c) * 9 + tap];
      }
    }
  }
#pragma unroll
  for (int o = 0; o < CO; ++o)
    off[((size_t)(b * CO + o) * H + y) * W + xcol] = acc[o];
}

// ---------------------------------------------------------------------------
// Deform conv: async-prefetch offsets to LDS, bilinear-sample im2col tile
// into LDS (bf16), then WMMA GEMM.  M=32 px, N=64, K=576.  1 wave = 1 tile.
// ---------------------------------------------------------------------------
__global__ __launch_bounds__(256) void k_deform(
    const float* __restrict__ h, const float* __restrict__ off,
    const __bf16* __restrict__ Bd, float* __restrict__ out) {
  constexpr int C = 64, H = 256, W = 256, K2 = 9, KD = 576;
  __shared__ alignas(32) __bf16 As[32 * KD];        // 36 KB
  __shared__ alignas(16) float Soff[18 * 32];       // 2.3 KB offset slab
  const int b = blockIdx.x >> 11;                   // 2048 blocks per image
  const int P0 = (blockIdx.x & 2047) * 32;          // 32 pixels, same row
  const int t = threadIdx.x;

#ifdef HAVE_ASYNC
  // ASYNCcnt-tracked global->LDS copy of this block's 18x32 offset slab.
  if (t < 144) {
    const int ch = t >> 3, chunk = t & 7;
    const float* gsrc = off + ((size_t)(b * 18 + ch) << 16) + P0 + chunk * 4;
    __builtin_amdgcn_global_load_async_to_lds_b128(
        (g_i32x4*)gsrc, (l_i32x4*)&Soff[ch * 32 + chunk * 4], 0, 0);
  }
  __builtin_amdgcn_s_wait_asynccnt(0);
  __syncthreads();
#else
  if (t < 576) {
    const int ch = t >> 5, pp = t & 31;
    Soff[ch * 32 + pp] = off[((size_t)(b * 18 + ch) << 16) + P0 + pp];
  }
  __syncthreads();
#endif

  // phase 1: bilinear gather of the 32x9x64 im2col tile
  for (int it = t; it < 32 * K2; it += 256) {
    const int p = it / K2, k = it % K2;
    const int P = P0 + p, y = P >> 8, xcol = P & 255;
    const float dyo = Soff[(2 * k) * 32 + p];
    const float dxo = Soff[(2 * k + 1) * 32 + p];
    const float py = (float)y + (float)(k / 3 - 1) + dyo;
    const float px = (float)xcol + (float)(k % 3 - 1) + dxo;
    const float y0f = floorf(py), x0f = floorf(px);
    float wgt[4];
    int idx4[4];
#pragma unroll
    for (int tap = 0; tap < 4; ++tap) {
      const float yy = y0f + (float)(tap >> 1);
      const float xx = x0f + (float)(tap & 1);
      const float wv = (1.0f - fabsf(py - yy)) * (1.0f - fabsf(px - xx));
      const bool valid =
          (yy >= 0.f) & (yy < (float)H) & (xx >= 0.f) & (xx < (float)W);
      wgt[tap] = valid ? wv : 0.0f;
      const int yc = (int)fminf(fmaxf(yy, 0.f), (float)(H - 1));
      const int xc = (int)fminf(fmaxf(xx, 0.f), (float)(W - 1));
      idx4[tap] = yc * W + xc;
    }
    const float* hb = h + (size_t)b * C * H * W;
    __bf16* dst = &As[p * KD + k * C];
    for (int c = 0; c < C; ++c) {
      const float* hc = hb + (size_t)c * H * W;
      const float s = wgt[0] * hc[idx4[0]] + wgt[1] * hc[idx4[1]] +
                      wgt[2] * hc[idx4[2]] + wgt[3] * hc[idx4[3]];
      dst[c] = (__bf16)s;
    }
  }
  __syncthreads();

  // phase 2: WMMA GEMM, one 16x16 tile per wave, 18 k-steps
  const int wave = t >> 5, lane = t & 31;
  const int lrow = lane & 15;
  const int khalf = (lane >> 4) << 3, kh16 = (lane >> 4) << 4;
  const int mt = wave >> 2, nt = wave & 3;
  const int arow = mt * 16 + lrow;
  const int bcol = nt * 16 + lrow;
  v8f acc = {};
  for (int k0 = 0; k0 < KD; k0 += 32) {
    v16bf a = *(const v16bf*)(&As[arow * KD + k0 + kh16]);
    v16bf bm = *(const v16bf*)(&Bd[bcol * KD + k0 + kh16]);
    acc = __builtin_amdgcn_wmma_f32_16x16x32_bf16(false, a, false, bm, (short)0,
                                                  acc, false, false);
  }
  const int n = nt * 16 + lrow;
  const int mbase = mt * 16 + ((lane >> 4) << 3);
#pragma unroll
  for (int r = 0; r < 8; ++r) {
    out[(size_t)(b * C + n) * (H * W) + P0 + mbase + r] = acc[r];
  }
}

// ---------------------------------------------------------------------------
// Batch-stat reduction + ABN (batchnorm with batch stats + leaky relu)
// ---------------------------------------------------------------------------
__global__ void k_zero(float* __restrict__ p, int n) {
  const int i = blockIdx.x * blockDim.x + threadIdx.x;
  if (i < n) p[i] = 0.0f;
}

__global__ __launch_bounds__(256) void k_stats(const float* __restrict__ v,
                                               float* __restrict__ stats) {
  __shared__ float ssum[256], ssq[256];
  const size_t base = (size_t)blockIdx.x * 4096;    // contiguous, single chan
  const int c = (int)((base >> 16) & 63);
  float s = 0.f, q = 0.f;
#pragma unroll 4
  for (int i = 0; i < 16; ++i) {
    const float a = v[base + i * 256 + threadIdx.x];
    s += a;
    q += a * a;
  }
  ssum[threadIdx.x] = s;
  ssq[threadIdx.x] = q;
  __syncthreads();
  for (int d = 128; d > 0; d >>= 1) {
    if (threadIdx.x < (unsigned)d) {
      ssum[threadIdx.x] += ssum[threadIdx.x + d];
      ssq[threadIdx.x] += ssq[threadIdx.x + d];
    }
    __syncthreads();
  }
  if (threadIdx.x == 0) {
    atomicAdd(&stats[c], ssum[0]);
    atomicAdd(&stats[64 + c], ssq[0]);
  }
}

__global__ __launch_bounds__(256) void k_abn(const float* __restrict__ v,
                                             const float* __restrict__ stats,
                                             const float* __restrict__ g,
                                             const float* __restrict__ bb,
                                             float* __restrict__ out) {
  const size_t e0 = ((size_t)blockIdx.x * 256 + threadIdx.x) * 4;
  const int c = (int)((e0 >> 16) & 63);
  const float invN = 1.0f / 131072.0f;              // B*H*W
  const float mean = stats[c] * invN;
  const float var = stats[64 + c] * invN - mean * mean;
  const float sc = rsqrtf(var + 1e-5f) * g[c];
  const float sh = bb[c];
#pragma unroll
  for (int i = 0; i < 4; ++i) {
    const float y = (v[e0 + i] - mean) * sc + sh;
    out[e0 + i] = (y >= 0.f) ? y : 0.01f * y;
  }
}

// ---------------------------------------------------------------------------
extern "C" void kernel_launch(void* const* d_in, const int* in_sizes, int n_in,
                              void* d_out, int out_size, void* d_ws,
                              size_t ws_size, hipStream_t stream) {
  const float* x     = (const float*)d_in[0];
  const float* dw_w  = (const float*)d_in[1];
  const float* dw_b  = (const float*)d_in[2];
  const float* pw_w  = (const float*)d_in[3];
  const float* pw_b  = (const float*)d_in[4];
  const float* off1w = (const float*)d_in[5];
  const float* off1b = (const float*)d_in[6];
  const float* dcn1w = (const float*)d_in[7];
  const float* bn1g  = (const float*)d_in[8];
  const float* bn1b  = (const float*)d_in[9];
  const float* off2w = (const float*)d_in[10];
  const float* off2b = (const float*)d_in[11];
  const float* dcn2w = (const float*)d_in[12];
  const float* bn2g  = (const float*)d_in[13];
  const float* bn2b  = (const float*)d_in[14];

  char* ws = (char*)d_ws;
  size_t o = 0;
  float* bufA = (float*)(ws + o); o += (size_t)8388608 * 4;   // [2,64,256,256]
  float* bufB = (float*)(ws + o); o += (size_t)8388608 * 4;
  float* bufO = (float*)(ws + o); o += (size_t)2359296 * 4;   // [2,18,256,256]
  __bf16* Bpw = (__bf16*)(ws + o); o += (size_t)4096 * 2;
  __bf16* Bd1 = (__bf16*)(ws + o); o += (size_t)36864 * 2;
  __bf16* Bd2 = (__bf16*)(ws + o); o += (size_t)36864 * 2;
  float* stats = (float*)(ws + o); o += 128 * 4;

  // weight prep (bf16, B-matrix layout)
  k_prep_pw<<<16, 256, 0, stream>>>(pw_w, Bpw);
  k_prep_dcn<<<144, 256, 0, stream>>>(dcn1w, Bd1);
  k_prep_dcn<<<144, 256, 0, stream>>>(dcn2w, Bd2);

  // depthwise + pointwise (WMMA, TDM-staged B matrix)
  k_dw_pw<<<dim3(16, 16, 2), 256, 0, stream>>>(x, dw_w, dw_b, Bpw, pw_b, bufA);

  // stage 1: offsets -> deform conv (WMMA) -> ABN
  k_off<<<512, 256, 0, stream>>>(bufA, off1w, off1b, bufO);
  k_deform<<<4096, 256, 0, stream>>>(bufA, bufO, Bd1, bufB);
  k_zero<<<1, 128, 0, stream>>>(stats, 128);
  k_stats<<<2048, 256, 0, stream>>>(bufB, stats);
  k_abn<<<8192, 256, 0, stream>>>(bufB, stats, bn1g, bn1b, bufA);

  // stage 2
  k_off<<<512, 256, 0, stream>>>(bufA, off2w, off2b, bufO);
  k_deform<<<4096, 256, 0, stream>>>(bufA, bufO, Bd2, bufB);
  k_zero<<<1, 128, 0, stream>>>(stats, 128);
  k_stats<<<2048, 256, 0, stream>>>(bufB, stats);
  k_abn<<<8192, 256, 0, stream>>>(bufB, stats, bn2g, bn2b, (float*)d_out);
}
